// LayerAdapter_25855703122320
// MI455X (gfx1250) — matrix operand order repaired and checked by reading, not compiled, not run on hardware
//
#include <hip/hip_runtime.h>
#include <hip/hip_bf16.h>
#include <math.h>

// ---------------- problem constants (from reference) ----------------
#define BB   4
#define TT   512
#define BT   2048          // BB*TT tokens
#define DD   1024          // hidden
#define MM   4096          // kv dim
#define LL   24            // layers
#define HH   4             // heads
#define HD   8             // head qk dim
#define HM   16384         // HH*MM
#define LN_EPS 1e-5f
#define SCOREF 0.17677669529663687f   // 1/(sqrt(8)*2)

typedef __bf16 v16bf __attribute__((ext_vector_type(16)));
typedef __bf16 v8bf  __attribute__((ext_vector_type(8)));
typedef float  v8f   __attribute__((ext_vector_type(8)));

// =====================================================================
// Kernel 1: one workgroup per token. Streams c_fc twice (2nd pass is
// L2-resident), produces out (H,M) in bf16 into workspace.
// =====================================================================
__global__ __launch_bounds__(256)
void la_token_kernel(const float* __restrict__ base,
                     const float* __restrict__ cfc,
                     const float* __restrict__ qg,  const float* __restrict__ qb_,
                     const float* __restrict__ kg,  const float* __restrict__ kb,
                     const float* __restrict__ Wq,  const float* __restrict__ bq,
                     const float* __restrict__ Wk,  const float* __restrict__ bk,
                     __bf16* __restrict__ out_ws)
{
    __shared__ __bf16 cg[HH * MM];          // g[m] * c_h[m], bf16 (32 KB)
    __shared__ float  Qn[DD];               // normalized Q row
    __shared__ float  red[256];             // generic reduction / score scratch
    __shared__ float  Qlow[32];
    __shared__ float  qdotb[HH];
    __shared__ float  Gh[HH], Bc[HH];
    __shared__ float  rowstats[LL][6];      // per-l: sum,sumsq,A0..A3 -> mu,rs
    __shared__ float  wl[HH][LL];           // attn * rstd
    __shared__ float  sh[HH];               // sum_l attn*rstd*mu

    const int tok  = blockIdx.x;
    const int tid  = threadIdx.x;
    const int lane = tid & 31;
    const int wave = tid >> 5;

    const float* x  = base + (size_t)tok * DD;
    const float* kv = cfc  + (size_t)tok * MM * LL;   // [m][l] contiguous

    // ---------- P0: layernorm of base row ----------
    {
        float s = 0.f, sq = 0.f;
        for (int j = tid; j < DD; j += 256) { float v = x[j]; s += v; sq += v * v; }
        for (int o = 16; o > 0; o >>= 1) { s += __shfl_down(s, o); sq += __shfl_down(sq, o); }
        if (lane == 0) { red[wave] = s; red[8 + wave] = sq; }
        __syncthreads();
        if (tid == 0) {
            float S = 0.f, SQ = 0.f;
            for (int w = 0; w < 8; ++w) { S += red[w]; SQ += red[8 + w]; }
            float mu = S / DD;
            float var = SQ / DD - mu * mu;
            red[16] = mu; red[17] = rsqrtf(var + LN_EPS);
        }
        __syncthreads();
        float mu = red[16], rs = red[17];
        for (int j = tid; j < DD; j += 256)
            Qn[j] = (x[j] - mu) * rs * qg[j] + qb_[j];
        __syncthreads();
    }

    // ---------- P0.5: Q_low = Qn @ Wq.T + bq ----------
    {
        int i = tid & 31, seg = tid >> 5;
        float p = 0.f;
        const float* wr = Wq + (size_t)i * DD;
        for (int j = seg * 128; j < seg * 128 + 128; ++j) p += Qn[j] * wr[j];
        red[seg * 32 + i] = p;
    }
    __syncthreads();
    if (tid < 32) {
        float p = 0.f;
        for (int g = 0; g < 8; ++g) p += red[g * 32 + tid];
        Qlow[tid] = p + bq[tid];
    }
    __syncthreads();
    if (tid < HH) {
        float p = 0.f;
        for (int d = 0; d < HD; ++d) p += Qlow[tid * HD + d] * bk[tid * HD + d];
        qdotb[tid] = p;
    }

    // ---------- P1: cg[h][m] = g[m]*c_h[m];  Gh = sum cg;  Bc = sum b*c ----------
    {
        float ql[32];
        #pragma unroll
        for (int i = 0; i < 32; ++i) ql[i] = Qlow[i];
        float Ga[HH] = {0.f, 0.f, 0.f, 0.f}, Ba[HH] = {0.f, 0.f, 0.f, 0.f};
        for (int m = tid; m < MM; m += 256) {
            float gm = kg[m], bm = kb[m];
            #pragma unroll
            for (int h = 0; h < HH; ++h) {
                float c = 0.f;
                #pragma unroll
                for (int d = 0; d < HD; ++d)
                    c += ql[h * HD + d] * Wk[(size_t)(h * HD + d) * MM + m];
                float cv = gm * c;
                cg[h * MM + m] = (__bf16)cv;
                Ga[h] += cv;
                Ba[h] += bm * c;
            }
        }
        #pragma unroll
        for (int h = 0; h < HH; ++h)
            for (int o = 16; o > 0; o >>= 1) {
                Ga[h] += __shfl_down(Ga[h], o);
                Ba[h] += __shfl_down(Ba[h], o);
            }
        if (lane == 0) {
            #pragma unroll
            for (int h = 0; h < HH; ++h) {
                red[wave * 4 + h]      = Ga[h];
                red[32 + wave * 4 + h] = Ba[h];
            }
        }
        __syncthreads();
        if (tid < HH) {
            float G = 0.f, Bv = 0.f;
            for (int w = 0; w < 8; ++w) { G += red[w * 4 + tid]; Bv += red[32 + w * 4 + tid]; }
            Gh[tid] = G; Bc[tid] = Bv;
        }
        __syncthreads();
    }

    // ---------- P2 (pass A over c_fc): per-row stats + head dots ----------
    {
        float a[3][6];
        #pragma unroll
        for (int r = 0; r < 3; ++r)
            #pragma unroll
            for (int j = 0; j < 6; ++j) a[r][j] = 0.f;

        for (int m = lane; m < MM; m += 32) {
            const float* bp = kv + (size_t)m * LL;
            // speculative prefetch ~12 KB ahead (OOB dropped silently)
            __builtin_prefetch(bp + 128 * LL, 0, 3);
            float c0 = (float)cg[0 * MM + m];
            float c1 = (float)cg[1 * MM + m];
            float c2 = (float)cg[2 * MM + m];
            float c3 = (float)cg[3 * MM + m];
            #pragma unroll
            for (int r = 0; r < 3; ++r) {
                float v = bp[wave + r * 8];        // rows l = wave, wave+8, wave+16
                a[r][0] += v;      a[r][1] += v * v;
                a[r][2] += v * c0; a[r][3] += v * c1;
                a[r][4] += v * c2; a[r][5] += v * c3;
            }
        }
        #pragma unroll
        for (int r = 0; r < 3; ++r)
            #pragma unroll
            for (int j = 0; j < 6; ++j)
                for (int o = 16; o > 0; o >>= 1) a[r][j] += __shfl_down(a[r][j], o);
        if (lane == 0) {
            #pragma unroll
            for (int r = 0; r < 3; ++r)
                #pragma unroll
                for (int j = 0; j < 6; ++j) rowstats[wave + r * 8][j] = a[r][j];
        }
        __syncthreads();
    }

    // ---------- P3: scores (+NaN mask) and softmax ----------
    if (tid < LL) {
        int l = tid;
        float S = rowstats[l][0], SQ = rowstats[l][1];
        float mu = S / MM;
        float var = SQ / MM - mu * mu;
        float rs = rsqrtf(var + LN_EPS);
        #pragma unroll
        for (int h = 0; h < HH; ++h) {
            float A = rowstats[l][2 + h];
            float scr = (rs * (A - mu * Gh[h]) + Bc[h] + qdotb[h]) * SCOREF;
            if (scr != scr) scr = -__builtin_inff();   // NaN -> -inf
            red[h * LL + l] = scr;
        }
        rowstats[l][0] = mu; rowstats[l][1] = rs;
    }
    __syncthreads();
    if (tid < HH) {
        int h = tid;
        float mx = -__builtin_inff();
        #pragma unroll
        for (int l = 0; l < LL; ++l) mx = fmaxf(mx, red[h * LL + l]);
        float se = 0.f;
        #pragma unroll
        for (int l = 0; l < LL; ++l) se += __expf(red[h * LL + l] - mx);
        float inv = 1.f / se;
        float sa = 0.f;
        #pragma unroll
        for (int l = 0; l < LL; ++l) {
            float at = __expf(red[h * LL + l] - mx) * inv;
            float w = at * rowstats[l][1];
            wl[h][l] = w;
            sa += w * rowstats[l][0];
        }
        sh[h] = sa;
    }
    __syncthreads();

    // ---------- P4 (pass B over c_fc): out[h][m], bf16 store ----------
    {
        float s0 = sh[0], s1 = sh[1], s2 = sh[2], s3 = sh[3];
        #pragma unroll 1
        for (int mi = 0; mi < 16; ++mi) {
            int m = tid + 256 * mi;
            const float4* p4 = (const float4*)(kv + (size_t)m * LL);
            // prefetch next chunk of this lane's stream (~24 KB ahead)
            __builtin_prefetch((const float*)p4 + 256 * LL, 0, 3);
            float4 q0 = p4[0], q1 = p4[1], q2 = p4[2], q3 = p4[3], q4 = p4[4], q5 = p4[5];
            float v[24] = { q0.x,q0.y,q0.z,q0.w, q1.x,q1.y,q1.z,q1.w,
                            q2.x,q2.y,q2.z,q2.w, q3.x,q3.y,q3.z,q3.w,
                            q4.x,q4.y,q4.z,q4.w, q5.x,q5.y,q5.z,q5.w };
            float a0 = 0.f, a1 = 0.f, a2 = 0.f, a3 = 0.f;
            #pragma unroll
            for (int l = 0; l < LL; ++l) {
                float vv = v[l];
                a0 += wl[0][l] * vv; a1 += wl[1][l] * vv;
                a2 += wl[2][l] * vv; a3 += wl[3][l] * vv;
            }
            float gm = kg[m], bm = kb[m];
            size_t o = (size_t)tok * HM + m;
            out_ws[o + 0 * MM] = (__bf16)(gm * (a0 - s0) + bm);
            out_ws[o + 1 * MM] = (__bf16)(gm * (a1 - s1) + bm);
            out_ws[o + 2 * MM] = (__bf16)(gm * (a2 - s2) + bm);
            out_ws[o + 3 * MM] = (__bf16)(gm * (a3 - s3) + bm);
        }
    }
}

// =====================================================================
// Kernel 2: outB f32 -> bf16
// =====================================================================
__global__ __launch_bounds__(256)
void la_cvt_outB_kernel(const float* __restrict__ src, __bf16* __restrict__ dst, int n)
{
    int i = blockIdx.x * 256 + threadIdx.x;
    if (i < n) dst[i] = (__bf16)src[i];
}

// =====================================================================
// Kernel 3: U[2048,64] = out_bf16[2048,16384] @ outB_bf16[64,16384]^T
// WMMA bf16 16x16x32; 8 waves per WG: 4 n-tiles x 2 K-halves, LDS combine.
// A layout per ISA (16-bit A 16x32): lane half splits K 0-7/16-23 vs 8-15/24-31.
// B layout: col = lane&15, same K split per lane half as A.
// =====================================================================
__global__ __launch_bounds__(256)
void la_gemm_lowrank_kernel(const __bf16* __restrict__ A,
                            const __bf16* __restrict__ Bm,
                            float* __restrict__ U)
{
    __shared__ float part[4][16][16];
    const int lane  = threadIdx.x & 31;
    const int wave  = threadIdx.x >> 5;
    const int ntile = wave & 3;
    const int khalf = wave >> 2;
    const int t0    = blockIdx.x * 16;
    const int n0    = ntile * 16;
    const int row   = lane & 15;
    const int half  = lane >> 4;

    const __bf16* arow = A  + (size_t)(t0 + row) * HM;
    const __bf16* brow = Bm + (size_t)(n0 + row) * HM;

    v8f acc = {};
    const int kbeg = khalf * (HM / 2);
    for (int k0 = kbeg; k0 < kbeg + HM / 2; k0 += 32) {
        // prefetch the A stream (B is L2-resident, 2 MB shared by all WGs)
        __builtin_prefetch(arow + k0 + 512, 0, 3);
        v8bf alo = *(const v8bf*)(arow + k0 + half * 8);        // K half*8..+7
        v8bf ahi = *(const v8bf*)(arow + k0 + 16 + half * 8);   // K 16+half*8..+7
        v16bf av = __builtin_shufflevector(alo, ahi,
                     0,1,2,3,4,5,6,7,8,9,10,11,12,13,14,15);
        v8bf blo = *(const v8bf*)(brow + k0 + half * 8);        // K half*8..+7
        v8bf bhi = *(const v8bf*)(brow + k0 + 16 + half * 8);   // K 16+half*8..+7
        v16bf bv = __builtin_shufflevector(blo, bhi,
                     0,1,2,3,4,5,6,7,8,9,10,11,12,13,14,15);
        acc = __builtin_amdgcn_wmma_f32_16x16x32_bf16(
                  false, av, false, bv, (short)0, acc, false, false);
    }
    if (khalf == 1) {
        #pragma unroll
        for (int j = 0; j < 8; ++j) part[ntile][j + 8 * half][row] = acc[j];
    }
    __syncthreads();
    if (khalf == 0) {
        #pragma unroll
        for (int j = 0; j < 8; ++j) {
            int mrow = j + 8 * half;
            U[(size_t)(t0 + mrow) * 64 + n0 + row] = acc[j] + part[ntile][mrow][row];
        }
    }
}

// =====================================================================
// Kernel 4: final = base + scale * (U @ outA.T + out_bias)
// =====================================================================
__global__ __launch_bounds__(256)
void la_final_kernel(const float* __restrict__ base,
                     const float* __restrict__ U,
                     const float* __restrict__ outA,
                     const float* __restrict__ obias,
                     const float* __restrict__ scale_p,
                     float* __restrict__ out)
{
    int idx = blockIdx.x * 256 + threadIdx.x;      // 0 .. BT*DD-1
    int i   = idx & (DD - 1);
    int tok = idx >> 10;                           // DD = 1024
    const float* u = U    + (size_t)tok * 64;
    const float* a = outA + (size_t)i * 64;
    float acc = 0.f;
    #pragma unroll
    for (int r = 0; r < 64; ++r) acc += u[r] * a[r];
    out[idx] = base[idx] + scale_p[0] * (acc + obias[i]);
}

// =====================================================================
extern "C" void kernel_launch(void* const* d_in, const int* in_sizes, int n_in,
                              void* d_out, int out_size, void* d_ws, size_t ws_size,
                              hipStream_t stream)
{
    const float* base  = (const float*)d_in[0];
    const float* cfc   = (const float*)d_in[1];
    const float* qg    = (const float*)d_in[2];
    const float* qb    = (const float*)d_in[3];
    const float* kg    = (const float*)d_in[4];
    const float* kb    = (const float*)d_in[5];
    const float* Wq    = (const float*)d_in[6];
    const float* bq    = (const float*)d_in[7];
    const float* Wk    = (const float*)d_in[8];
    const float* bk    = (const float*)d_in[9];
    const float* outA  = (const float*)d_in[10];
    const float* outB  = (const float*)d_in[11];
    const float* obias = (const float*)d_in[12];
    const float* scal  = (const float*)d_in[13];
    float* out = (float*)d_out;

    // workspace layout (all regions fully overwritten each call):
    //   [0, 64MB)          out bf16  (BT x HM)
    //   [64MB, 66MB)       outB bf16 (64 x HM)
    //   [66MB, 66.5MB)     U f32     (BT x 64)
    const size_t OUT_WS_B  = (size_t)BT * HM * sizeof(__bf16);   // 67,108,864
    const size_t OUTB_WS_B = (size_t)64 * HM * sizeof(__bf16);   //  2,097,152
    __bf16* out_ws  = (__bf16*)d_ws;
    __bf16* outB_bf = (__bf16*)((char*)d_ws + OUT_WS_B);
    float*  U       = (float*)((char*)d_ws + OUT_WS_B + OUTB_WS_B);

    la_token_kernel<<<BT, 256, 0, stream>>>(base, cfc, qg, qb, kg, kb,
                                            Wq, bq, Wk, bk, out_ws);
    la_cvt_outB_kernel<<<(64 * HM + 255) / 256, 256, 0, stream>>>(outB, outB_bf, 64 * HM);
    la_gemm_lowrank_kernel<<<BT / 16, 256, 0, stream>>>(out_ws, outB_bf, U);
    la_final_kernel<<<(BT * DD) / 256, 256, 0, stream>>>(base, U, outA, obias, scal, out);
}
